// MambaHART_57432302682156
// MI455X (gfx1250) — matrix operand order repaired and checked by reading, not compiled, not run on hardware
//
#include <hip/hip_runtime.h>
#include <hip/hip_bf16.h>
#include <cstdint>

// Problem constants (match reference)
#define B_SZ   4
#define SEQQ   2048
#define DMODEL 512
#define DINNER 1024
#define NSTATE 16
#define KCONV  17
#define MTOT   (B_SZ * SEQQ)   // 8192 rows for all GEMMs
#define NCHUNK 16
#define CLEN   (SEQQ / NCHUNK) // 128

typedef __attribute__((ext_vector_type(16))) _Float16 v16h;
typedef __attribute__((ext_vector_type(8)))  _Float16 v8h;
typedef __attribute__((ext_vector_type(4)))  _Float16 v4h;
typedef __attribute__((ext_vector_type(8)))  float    v8f;
typedef __attribute__((ext_vector_type(4)))  int      v4i;

// CDNA5 async global->LDS copy path (ASYNCcnt), guarded so the kernel still
// compiles cleanly if this toolchain doesn't expose the builtins.
#if defined(__has_builtin)
#  if __has_builtin(__builtin_amdgcn_global_load_async_to_lds_b128)
#    define HAVE_ASYNC_COPY 1
#  endif
#  if __has_builtin(__builtin_amdgcn_s_wait_asynccnt)
#    define HAVE_ASYNC_WAIT 1
#  endif
#endif
#ifndef HAVE_ASYNC_COPY
#  define HAVE_ASYNC_COPY 0
#endif
#ifndef HAVE_ASYNC_WAIT
#  define HAVE_ASYNC_WAIT 0
#endif
#define AS1 __attribute__((address_space(1)))
#define AS3 __attribute__((address_space(3)))

// ---------------------------------------------------------------------------
// Weight prep: fp32 (K x N) row-major -> f16 (N x K) row-major (transposed)
// ---------------------------------------------------------------------------
__global__ void wt_kernel(const float* __restrict__ W, _Float16* __restrict__ Wt,
                          int K, int N) {
    int gid = blockIdx.x * 256 + threadIdx.x;
    if (gid >= K * N) return;
    int k = gid / N, n = gid % N;
    Wt[(size_t)n * K + k] = (_Float16)W[gid];
}

// ---------------------------------------------------------------------------
// WMMA GEMM: OUT = epilogue( X(M x KTOT) * W(KTOT x NN) + bias )
//  MODE 0: silu, MODE 1: xc * sigmoid(.), MODE 2: identity
// Block tile 128x64, 8 waves of 32x32 (2x2 v_wmma_f32_16x16x32_f16).
// LDS double-buffered: stage tile k+1 (A: VGPR cvt path, B: async-to-LDS)
// while computing tile k; one barrier per K-step.
// ---------------------------------------------------------------------------
template <int MODE, int KTOT, int NN>
__global__ __launch_bounds__(256)
void gemm_kernel(const float* __restrict__ X, const _Float16* __restrict__ Wt,
                 const float* __restrict__ bias, const float* __restrict__ XC,
                 float* __restrict__ OUT) {
    constexpr int PA = 40;  // LDS pitch in halves (80B rows, 16B aligned)
    __shared__ __align__(16) _Float16 As[2][128 * PA];
    __shared__ __align__(16) _Float16 Bs[2][64 * PA];

    const int tid  = threadIdx.x;
    const int m0   = blockIdx.y * 128;
    const int n0   = blockIdx.x * 64;
    const int wid  = tid >> 5;
    const int lane = tid & 31;
    const int wm   = wid & 3;   // 4 row-groups of 32
    const int wn   = wid >> 2;  // 2 col-groups of 32
    const int hlf  = lane >> 4;
    const int l16  = lane & 15;

    // A-tile fill: thread covers 16 consecutive floats of the 128x32 tile
    const int ar  = tid >> 1;
    const int acb = (tid & 1) * 16;
    // B-tile fill: thread covers 8 consecutive halves (16B) of the 64x32 tile
    const int bn = tid >> 2;
    const int bc = (tid & 3) * 8;

    const float*    xrow = X + (size_t)(m0 + ar) * KTOT + acb;
    const _Float16* wrow = Wt + (size_t)(n0 + bn) * KTOT + bc;

    v8f acc[2][2] = {};

    auto stageA = [&](int buf, int k0) {
#pragma unroll
        for (int j = 0; j < 4; ++j) {
            const float4 v = *(const float4*)(xrow + k0 + j * 4);
            v4h hv = {(_Float16)v.x, (_Float16)v.y, (_Float16)v.z, (_Float16)v.w};
            *(v4h*)(&As[buf][ar * PA + acb + j * 4]) = hv;
        }
    };
    auto stageB = [&](int buf, int k0) {
#if HAVE_ASYNC_COPY
        __builtin_amdgcn_global_load_async_to_lds_b128(
            (AS1 v4i*)(wrow + k0),
            (AS3 v4i*)(&Bs[buf][bn * PA + bc]), 0, 0);
#else
        *(v8h*)(&Bs[buf][bn * PA + bc]) = *(const v8h*)(wrow + k0);
#endif
    };

    // prologue: stage tile 0
    stageB(0, 0);
    stageA(0, 0);
#if HAVE_ASYNC_WAIT
    __builtin_amdgcn_s_wait_asynccnt(0);
#endif
    __syncthreads();

    for (int k0 = 0; k0 < KTOT; k0 += 32) {
        const int cur = (k0 >> 5) & 1;
        const int nxt = cur ^ 1;

        // stage next tile while computing on current one
        if (k0 + 32 < KTOT) {
            stageB(nxt, k0 + 32);
            stageA(nxt, k0 + 32);
        }
        if (k0 + 64 < KTOT) {  // gfx1250 global_prefetch two tiles ahead
            __builtin_prefetch(xrow + k0 + 64, 0, 3);
            __builtin_prefetch(wrow + k0 + 64, 0, 3);
        }

        v16h af[2], bf[2];
#pragma unroll
        for (int mi = 0; mi < 2; ++mi) {
            int row = wm * 32 + mi * 16 + l16;
            v8h lo = *(const v8h*)(&As[cur][row * PA + hlf * 8]);
            v8h hi = *(const v8h*)(&As[cur][row * PA + 16 + hlf * 8]);
#pragma unroll
            for (int e = 0; e < 8; ++e) { af[mi][e] = lo[e]; af[mi][8 + e] = hi[e]; }
        }
#pragma unroll
        for (int ni = 0; ni < 2; ++ni) {
            int col = wn * 32 + ni * 16 + l16;
            v8h lo = *(const v8h*)(&Bs[cur][col * PA + hlf * 8]);
            v8h hi = *(const v8h*)(&Bs[cur][col * PA + 16 + hlf * 8]);
#pragma unroll
            for (int e = 0; e < 8; ++e) { bf[ni][e] = lo[e]; bf[ni][8 + e] = hi[e]; }
        }
#pragma unroll
        for (int mi = 0; mi < 2; ++mi)
#pragma unroll
            for (int ni = 0; ni < 2; ++ni)
                acc[mi][ni] = __builtin_amdgcn_wmma_f32_16x16x32_f16(
                    false, af[mi], false, bf[ni], (short)0, acc[mi][ni], false, false);

#if HAVE_ASYNC_WAIT
        __builtin_amdgcn_s_wait_asynccnt(0);
#endif
        __syncthreads();
    }

    // --- fused epilogue (C/D layout: lane L, elem i -> row=(L/16)*8+i, col=L%16)
#pragma unroll
    for (int mi = 0; mi < 2; ++mi)
#pragma unroll
        for (int ni = 0; ni < 2; ++ni) {
            int col = n0 + wn * 32 + ni * 16 + l16;
            float bb = bias[col];
#pragma unroll
            for (int i = 0; i < 8; ++i) {
                int row = m0 + wm * 32 + mi * 16 + hlf * 8 + i;
                float v = acc[mi][ni][i] + bb;
                float o;
                if (MODE == 0) {
                    o = v * (1.f / (1.f + __expf(-v)));               // silu
                } else if (MODE == 1) {
                    float xc = XC[(size_t)row * NN + col];
                    o = xc * (1.f / (1.f + __expf(-v)));              // gate
                } else {
                    o = v;
                }
                OUT[(size_t)row * NN + col] = o;
            }
        }
}

// ---------------------------------------------------------------------------
// Depthwise conv along t, K=17, SAME (pad 8). Thread = (b, d), 16 t-outputs.
// ---------------------------------------------------------------------------
__global__ __launch_bounds__(256)
void conv_kernel(const float* __restrict__ Xin, const float* __restrict__ CW,
                 const float* __restrict__ CB, float* __restrict__ Xc) {
    const int d  = blockIdx.x * 256 + threadIdx.x;
    const int t0 = blockIdx.y * 16;
    const int b  = blockIdx.z;
    float cw[KCONV];
#pragma unroll
    for (int k = 0; k < KCONV; ++k) cw[k] = CW[d * KCONV + k];
    const float cb = CB[d];
    float xwin[32];
#pragma unroll
    for (int j = 0; j < 32; ++j) {
        int tin = t0 + j - 8;
        xwin[j] = (tin >= 0 && tin < SEQQ)
                      ? Xin[((size_t)(b * SEQQ + tin)) * DINNER + d] : 0.f;
    }
#pragma unroll
    for (int i = 0; i < 16; ++i) {
        float a = cb;
#pragma unroll
        for (int k = 0; k < KCONV; ++k) a = fmaf(cw[k], xwin[i + k], a);
        Xc[((size_t)(b * SEQQ + t0 + i)) * DINNER + d] = a;
    }
}

// ---------------------------------------------------------------------------
// Chunked diagonal SSM scan (3 passes: local, carry, final+fused C-dot)
// ---------------------------------------------------------------------------
__global__ __launch_bounds__(256)
void scan_local(const float* __restrict__ XG, const float* __restrict__ Alog,
                const float* __restrict__ Bp, float* __restrict__ Hfin) {
    const int d = blockIdx.x * 256 + threadIdx.x;
    const int c = blockIdx.y, b = blockIdx.z;
    float A[NSTATE], Bc[NSTATE], h[NSTATE];
#pragma unroll
    for (int n = 0; n < NSTATE; ++n) {
        A[n]  = -__expf(Alog[d * NSTATE + n]);
        Bc[n] = Bp[d * NSTATE + n];
        h[n]  = 0.f;
    }
    const float* xp = XG + ((size_t)(b * SEQQ + c * CLEN)) * DINNER + d;
    for (int t = 0; t < CLEN; ++t) {
        float u = xp[(size_t)t * DINNER];
#pragma unroll
        for (int n = 0; n < NSTATE; ++n) h[n] = fmaf(A[n], h[n], u * Bc[n]);
    }
    float* hp = Hfin + ((size_t)((b * NCHUNK + c) * DINNER + d)) * NSTATE;
#pragma unroll
    for (int n = 0; n < NSTATE; ++n) hp[n] = h[n];
}

__global__ void scan_carry(const float* __restrict__ Alog,
                           const float* __restrict__ Hfin,
                           float* __restrict__ Hin) {
    int gid = blockIdx.x * 256 + threadIdx.x;  // B*D*N = 65536 threads
    int n = gid & (NSTATE - 1);
    int d = (gid >> 4) & (DINNER - 1);
    int b = gid >> 14;
    // A^CLEN with CLEN even: (-exp(a))^128 = exp(128*a)
    float AL = __expf((float)CLEN * Alog[d * NSTATE + n]);
    float h = 0.f;
    for (int c = 0; c < NCHUNK; ++c) {
        size_t idx = ((size_t)((b * NCHUNK + c) * DINNER + d)) * NSTATE + n;
        Hin[idx] = h;
        h = fmaf(AL, h, Hfin[idx]);
    }
}

__global__ __launch_bounds__(256)
void scan_final(float* __restrict__ XG /* in: xg, out: y (in place) */,
                const float* __restrict__ Alog, const float* __restrict__ Bp,
                const float* __restrict__ Cp, const float* __restrict__ Dp,
                const float* __restrict__ Hin) {
    const int d = blockIdx.x * 256 + threadIdx.x;
    const int c = blockIdx.y, b = blockIdx.z;
    float A[NSTATE], Bc[NSTATE], Cc[NSTATE], h[NSTATE];
    const float* hp = Hin + ((size_t)((b * NCHUNK + c) * DINNER + d)) * NSTATE;
#pragma unroll
    for (int n = 0; n < NSTATE; ++n) {
        A[n]  = -__expf(Alog[d * NSTATE + n]);
        Bc[n] = Bp[d * NSTATE + n];
        Cc[n] = Cp[d * NSTATE + n];
        h[n]  = hp[n];
    }
    const float dd = Dp[d];
    float* xp = XG + ((size_t)(b * SEQQ + c * CLEN)) * DINNER + d;
    for (int t = 0; t < CLEN; ++t) {
        float u = xp[(size_t)t * DINNER];
        float y = 0.f;
#pragma unroll
        for (int n = 0; n < NSTATE; ++n) {
            h[n] = fmaf(A[n], h[n], u * Bc[n]);
            y    = fmaf(h[n], Cc[n], y);
        }
        xp[(size_t)t * DINNER] = y + dd * u;  // safe in-place: same element
    }
}

// ---------------------------------------------------------------------------
// Host-side orchestration (all on `stream`, graph-capture safe)
// ---------------------------------------------------------------------------
extern "C" void kernel_launch(void* const* d_in, const int* in_sizes, int n_in,
                              void* d_out, int out_size, void* d_ws, size_t ws_size,
                              hipStream_t stream) {
    (void)in_sizes; (void)n_in; (void)out_size; (void)ws_size;
    const float* inputs   = (const float*)d_in[0];
    const float* W_expand = (const float*)d_in[1];
    const float* b_expand = (const float*)d_in[2];
    const float* conv_w   = (const float*)d_in[3];
    const float* conv_b   = (const float*)d_in[4];
    const float* W_gate   = (const float*)d_in[5];
    const float* b_gate   = (const float*)d_in[6];
    const float* A_log    = (const float*)d_in[7];
    const float* Bp       = (const float*)d_in[8];
    const float* Cp       = (const float*)d_in[9];
    const float* Dp       = (const float*)d_in[10];
    const float* W_out    = (const float*)d_in[11];
    const float* b_out    = (const float*)d_in[12];
    float* out = (float*)d_out;

    // Workspace layout (~80MB)
    char*  ws   = (char*)d_ws;
    float* buf0 = (float*)(ws);                         // x -> xg -> y (in place)
    float* buf1 = (float*)(ws + 33554432ull);           // xc
    float* hfin = (float*)(ws + 67108864ull);
    float* hin  = (float*)(ws + 71303168ull);
    _Float16* wtE = (_Float16*)(ws + 75497472ull);      // 1024 x 512
    _Float16* wtG = (_Float16*)(ws + 76546048ull);      // 1024 x 1024
    _Float16* wtO = (_Float16*)(ws + 78643200ull);      // 512 x 1024

    // 1) weight transpose/convert to f16 (N-major)
    wt_kernel<<<(DMODEL * DINNER) / 256, 256, 0, stream>>>(W_expand, wtE, DMODEL, DINNER);
    wt_kernel<<<(DINNER * DINNER) / 256, 256, 0, stream>>>(W_gate, wtG, DINNER, DINNER);
    wt_kernel<<<(DINNER * DMODEL) / 256, 256, 0, stream>>>(W_out, wtO, DINNER, DMODEL);

    // 2) expand GEMM + SiLU
    gemm_kernel<0, DMODEL, DINNER>
        <<<dim3(DINNER / 64, MTOT / 128), 256, 0, stream>>>(inputs, wtE, b_expand, nullptr, buf0);

    // 3) depthwise conv
    conv_kernel<<<dim3(DINNER / 256, SEQQ / 16, B_SZ), 256, 0, stream>>>(buf0, conv_w, conv_b, buf1);

    // 4) gate GEMM
    gemm_kernel<1, DINNER, DINNER>
        <<<dim3(DINNER / 64, MTOT / 128), 256, 0, stream>>>(buf1, wtG, b_gate, buf1, buf0);

    // 5) chunked SSM scan (in place on buf0)
    scan_local<<<dim3(DINNER / 256, NCHUNK, B_SZ), 256, 0, stream>>>(buf0, A_log, Bp, hfin);
    scan_carry<<<(B_SZ * DINNER * NSTATE) / 256, 256, 0, stream>>>(A_log, hfin, hin);
    scan_final<<<dim3(DINNER / 256, NCHUNK, B_SZ), 256, 0, stream>>>(buf0, A_log, Bp, Cp, Dp, hin);

    // 6) output GEMM
    gemm_kernel<2, DINNER, DMODEL>
        <<<dim3(DMODEL / 64, MTOT / 128), 256, 0, stream>>>(buf0, wtO, b_out, nullptr, out);
}
